// GConvGRUModel2_13864154431958
// MI455X (gfx1250) — matrix operand order repaired and checked by reading, not compiled, not run on hardware
//
#include <hip/hip_runtime.h>
#include <hip/hip_bf16.h>
#include <math.h>

#define NN      20000
#define T_STEPS 4
#define EE      320000
#define LL      2
#define KC      5
#define DD      128
#define RT_PER_CHUNK 5   // row tiles per block (1250 tiles = 250 chunks * 5)

typedef __attribute__((ext_vector_type(16))) __bf16 v16bf;
typedef __attribute__((ext_vector_type(8)))  float  v8f;

// ---------------------------------------------------------------- utilities
__global__ void fill_kernel(float* __restrict__ p, int n, float v) {
    int i = blockIdx.x * blockDim.x + threadIdx.x;
    if (i < n) p[i] = v;
}

__global__ void scale_copy_kernel(float* __restrict__ dst, const float* __restrict__ src,
                                  int n, float s) {
    int i = blockIdx.x * blockDim.x + threadIdx.x;
    if (i < n) dst[i] = s * src[i];
}

// -------------------------------------------------- weight repack f32->bf16
// Target layout per (l,k): [tileC][kk(4)][lane(32)][slot(16)] bf16, matching
// the CDNA5 WMMA B-fragment (32x16): lane<16 -> K=slot (slot<8) / slot+8 (slot>=8), N=lane ; lane>=16 ->
// K=slot+8 (slot<8) / slot+16 (slot>=8), N=lane-16 (within each 32-deep K tile kk).
__global__ void pack_w_kernel(const float* __restrict__ Wsrc, __bf16* __restrict__ dst,
                              int nTileC, int gateBase, int total) {
    int idx = blockIdx.x * blockDim.x + threadIdx.x;
    if (idx >= total) return;
    int s     = idx & 15;
    int lane  = (idx >> 4) & 31;
    int kk    = (idx >> 9) & 3;
    int t     = idx >> 11;                 // tileC + nTileC*(k + KC*l)
    int tileC = t % nTileC;
    int k     = (t / nTileC) % KC;
    int l     = t / (nTileC * KC);
    int col   = tileC * 16 + (lane & 15);
    int g     = gateBase + (col >> 7);
    int cc    = col & 127;
    int kin   = kk * 32 + ((lane >> 4) << 3) + (s & 7) + ((s >> 3) << 4);
    dst[idx] = (__bf16)Wsrc[((size_t)((l * 3 + g) * KC + k)) * DD * DD + (size_t)kin * DD + cc];
}

// ------------------------------------------------------------- WMMA GEMM
// OUT[n x C] (+)= X[n x 128] @ W[128 x C].
// Block = 4 waves (128 thr). Each wave owns a 32-col strip with its 8 B
// fragments persistent in registers; the block streams RT_PER_CHUNK row
// tiles, staging each 16x128 A tile once through (double-buffered) LDS as
// bf16 fragments shared by all 4 waves.
__global__ __launch_bounds__(128) void gemm_wmma(const float* __restrict__ X,
                                                 const __bf16* __restrict__ Wp,
                                                 float* __restrict__ OUT,
                                                 int C, int beta) {
    __shared__ __bf16 ldsA[2][2048];       // [buf][(kk*32+lane)*16 + slot]

    const int tx  = threadIdx.x;           // lane 0..31
    const int w   = threadIdx.y;           // wave 0..3
    const int tid = w * 32 + tx;
    const int lkk = tid >> 5;              // K-tile this thread stages
    const int lL  = tid & 31;              // fragment lane it stages
    const int stR   = lL & 15;             // A row within tile
    const int stOff = (lL >> 4) * 8;       // K sub-offset for that lane

    const int colbase = blockIdx.y * 128 + w * 32;
    const int l15 = tx & 15;
    const int hi  = tx >> 4;

    // ---- persistent B fragments: 4 K-tiles x 2 col subtiles
    v16bf bfrag[4][2];
    for (int kk = 0; kk < 4; ++kk)
        for (int c = 0; c < 2; ++c) {
            const int tileC = (colbase >> 4) + c;
            bfrag[kk][c] = *(const v16bf*)(Wp + ((size_t)(tileC * 4 + kk) * 32 + tx) * 16);
        }

    for (int i = 0; i < RT_PER_CHUNK; ++i) {
        const int rowtile = blockIdx.x * RT_PER_CHUNK + i;
        const int row0 = rowtile * 16;
        const int buf = i & 1;

        // ---- cooperative stage: 16x128 f32 -> bf16 fragment layout in LDS
        {
            const float* xr = X + (size_t)(row0 + stR) * DD + lkk * 32 + stOff;
            float4 a0 = *(const float4*)(xr);
            float4 a1 = *(const float4*)(xr + 4);
            float4 a2 = *(const float4*)(xr + 16);
            float4 a3 = *(const float4*)(xr + 20);
            __bf16* d = &ldsA[buf][(lkk * 32 + lL) * 16];
            d[0]  = (__bf16)a0.x; d[1]  = (__bf16)a0.y; d[2]  = (__bf16)a0.z; d[3]  = (__bf16)a0.w;
            d[4]  = (__bf16)a1.x; d[5]  = (__bf16)a1.y; d[6]  = (__bf16)a1.z; d[7]  = (__bf16)a1.w;
            d[8]  = (__bf16)a2.x; d[9]  = (__bf16)a2.y; d[10] = (__bf16)a2.z; d[11] = (__bf16)a2.w;
            d[12] = (__bf16)a3.x; d[13] = (__bf16)a3.y; d[14] = (__bf16)a3.z; d[15] = (__bf16)a3.w;
        }
        if (i + 1 < RT_PER_CHUNK) {        // prefetch next row tile
            __builtin_prefetch(X + (size_t)(row0 + 16 + stR) * DD + lkk * 32 + stOff, 0, 3);
        }
        __syncthreads();

        v8f acc[2];
        if (beta) {
            for (int c = 0; c < 2; ++c)
                for (int r = 0; r < 8; ++r)
                    acc[c][r] = OUT[(size_t)(row0 + hi * 8 + r) * C + colbase + c * 16 + l15];
        } else {
            for (int c = 0; c < 2; ++c)
                for (int r = 0; r < 8; ++r)
                    acc[c][r] = 0.0f;
        }

        for (int kk = 0; kk < 4; ++kk) {
            v16bf a = *(const v16bf*)&ldsA[buf][(kk * 32 + tx) * 16];
            acc[0] = __builtin_amdgcn_wmma_f32_16x16x32_bf16(
                false, a, false, bfrag[kk][0], (short)0, acc[0], false, false);
            acc[1] = __builtin_amdgcn_wmma_f32_16x16x32_bf16(
                false, a, false, bfrag[kk][1], (short)0, acc[1], false, false);
        }

        for (int c = 0; c < 2; ++c)
            for (int r = 0; r < 8; ++r)
                OUT[(size_t)(row0 + hi * 8 + r) * C + colbase + c * 16 + l15] = acc[c][r];
    }
}

// -------------------------------------------------------------- graph norm
__global__ void deg_kernel(const int* __restrict__ src, const int* __restrict__ dst,
                           const float* __restrict__ w, float* __restrict__ deg) {
    int e = blockIdx.x * blockDim.x + threadIdx.x;
    if (e >= EE) return;
    int s = src[e];
    if (s == dst[e]) return;
    atomicAdd(&deg[s], w[e]);
}

__global__ void what_kernel(const int* __restrict__ src, const int* __restrict__ dst,
                            const float* __restrict__ w, const float* __restrict__ deg,
                            float* __restrict__ what) {
    int e = blockIdx.x * blockDim.x + threadIdx.x;
    if (e >= EE) return;
    int s = src[e], d = dst[e];
    float ww = (s == d) ? 0.0f : w[e];
    float ds = deg[s], dd = deg[d];
    float is = (ds > 0.0f) ? rsqrtf(fmaxf(ds, 1e-12f)) : 0.0f;
    float id = (dd > 0.0f) ? rsqrtf(fmaxf(dd, 1e-12f)) : 0.0f;
    what[e] = -is * ww * id;
}

// scatter: OUT[dst] += alpha * what * X[src]; 32 lanes per edge, float4 each
__global__ void prop_kernel(const int* __restrict__ src, const int* __restrict__ dst,
                            const float* __restrict__ what, const float* __restrict__ Xin,
                            float* __restrict__ OUT, float alpha) {
    int gid = blockIdx.x * blockDim.x + threadIdx.x;
    int e = gid >> 5;
    if (e >= EE) return;
    int f = (gid & 31) << 2;
    float w = alpha * what[e];
    float4 x = *(const float4*)(Xin + (size_t)src[e] * DD + f);
    float* o = OUT + (size_t)dst[e] * DD + f;
    atomicAdd(o + 0, w * x.x);
    atomicAdd(o + 1, w * x.y);
    atomicAdd(o + 2, w * x.z);
    atomicAdd(o + 3, w * x.w);
}

// ------------------------------------------------------------- GRU gates
__global__ void gate_zr_kernel(const float* __restrict__ Xacc, const float* __restrict__ Hacc,
                               const float* __restrict__ bx, const float* __restrict__ bh,
                               const float* __restrict__ h, float* __restrict__ Z,
                               float* __restrict__ HR) {
    int idx = blockIdx.x * blockDim.x + threadIdx.x;
    if (idx >= NN * DD) return;
    int n = idx >> 7;
    int c = idx & 127;
    float zi = Xacc[(size_t)n * 384 + c]       + bx[c]       + Hacc[(size_t)n * 256 + c]       + bh[c];
    float ri = Xacc[(size_t)n * 384 + 128 + c] + bx[128 + c] + Hacc[(size_t)n * 256 + 128 + c] + bh[128 + c];
    float z = 1.0f / (1.0f + __expf(-zi));
    float r = 1.0f / (1.0f + __expf(-ri));
    Z[idx]  = z;
    HR[idx] = h[idx] * r;
}

__global__ void gate_h_kernel(const float* __restrict__ Xacc, const float* __restrict__ Hh,
                              const float* __restrict__ bx, const float* __restrict__ bh,
                              const float* __restrict__ Z, float* __restrict__ h) {
    int idx = blockIdx.x * blockDim.x + threadIdx.x;
    if (idx >= NN * DD) return;
    int n = idx >> 7;
    int c = idx & 127;
    float hti = Xacc[(size_t)n * 384 + 256 + c] + bx[256 + c] + Hh[idx] + bh[256 + c];
    float ht = tanhf(hti);
    float z  = Z[idx];
    float hn = z * h[idx] + (1.0f - z) * ht;
    h[idx] = fmaxf(hn, 0.0f);
}

// ------------------------------------------------------------- output head
__global__ void head_kernel(const float* __restrict__ h1, const float* __restrict__ Wd,
                            const float* __restrict__ bd, float* __restrict__ out) {
    int n = blockIdx.x * blockDim.x + threadIdx.x;
    if (n >= NN) return;
    float logits[16];
    for (int j = 0; j < 16; ++j) logits[j] = bd[j];
    const float* hr = h1 + (size_t)n * DD;
    for (int k = 0; k < DD; ++k) {
        float hv = hr[k];
        for (int j = 0; j < 16; ++j) logits[j] += hv * Wd[k * 16 + j];
    }
    float m = logits[0];
    for (int j = 1; j < 16; ++j) m = fmaxf(m, logits[j]);
    float ssum = 0.0f;
    for (int j = 0; j < 16; ++j) ssum += __expf(logits[j] - m);
    float lse = m + __logf(ssum);
    for (int j = 0; j < 16; ++j) out[(size_t)n * 16 + j] = logits[j] - lse;
}

// ---------------------------------------------------------------- launcher
extern "C" void kernel_launch(void* const* d_in, const int* in_sizes, int n_in,
                              void* d_out, int out_size, void* d_ws, size_t ws_size,
                              hipStream_t stream) {
    (void)in_sizes; (void)n_in; (void)out_size; (void)ws_size;
    const int*   ei  = (const int*)d_in[0];   // [T][2][E]
    const float* ew  = (const float*)d_in[1]; // [T][E]
    const float* emb = (const float*)d_in[2]; // [N][128]
    const float* Wx  = (const float*)d_in[3]; // [L][3][K][128][128]
    const float* bx  = (const float*)d_in[4]; // [L][3][128]
    const float* Wh  = (const float*)d_in[5];
    const float* bh  = (const float*)d_in[6];
    const float* Wd  = (const float*)d_in[7]; // [128][16]
    const float* bd  = (const float*)d_in[8]; // [16]
    float* out = (float*)d_out;

    // ---- carve workspace
    char* wsp = (char*)d_ws;
    auto alloc = [&](size_t bytes) -> char* {
        char* p = wsp;
        wsp += (bytes + 255) & ~(size_t)255;
        return p;
    };
    const size_t ND = (size_t)NN * DD;
    float* h0    = (float*)alloc(ND * 4);
    float* h1    = (float*)alloc(ND * 4);
    float* B0    = (float*)alloc(ND * 4);
    float* B1    = (float*)alloc(ND * 4);
    float* B2    = (float*)alloc(ND * 4);
    float* Xacc  = (float*)alloc((size_t)NN * 384 * 4);
    float* Hacc  = (float*)alloc((size_t)NN * 256 * 4);
    float* Zbuf  = (float*)alloc(ND * 4);
    float* HRbuf = (float*)alloc(ND * 4);
    float* deg   = (float*)alloc((size_t)NN * 4);
    float* whatv = (float*)alloc((size_t)EE * 4);
    __bf16* WxP   = (__bf16*)alloc((size_t)LL * KC * 24 * 2048 * 2); // C=384
    __bf16* WhP01 = (__bf16*)alloc((size_t)LL * KC * 16 * 2048 * 2); // C=256
    __bf16* WhP2  = (__bf16*)alloc((size_t)LL * KC * 8  * 2048 * 2); // C=128

    const size_t sliceX  = 24 * 2048;  // bf16 elements per (l,k) slice
    const size_t slice01 = 16 * 2048;
    const size_t slice2  = 8 * 2048;

    // ---- weight repack (every call; deterministic)
    {
        int tot = LL * KC * 24 * 2048;
        pack_w_kernel<<<(tot + 255) / 256, 256, 0, stream>>>(Wx, WxP, 24, 0, tot);
        tot = LL * KC * 16 * 2048;
        pack_w_kernel<<<(tot + 255) / 256, 256, 0, stream>>>(Wh, WhP01, 16, 0, tot);
        tot = LL * KC * 8 * 2048;
        pack_w_kernel<<<(tot + 255) / 256, 256, 0, stream>>>(Wh, WhP2, 8, 2, tot);
    }

    const int nd = (int)ND;
    const int gb256 = (nd + 255) / 256;
    fill_kernel<<<gb256, 256, 0, stream>>>(h0, nd, 1.0f);
    fill_kernel<<<gb256, 256, 0, stream>>>(h1, nd, 1.0f);

    auto gemm = [&](const float* X, const __bf16* Wp, float* OUT, int C, int beta) {
        dim3 g(250, C / 128), b(32, 4, 1);
        gemm_wmma<<<g, b, 0, stream>>>(X, Wp, OUT, C, beta);
    };
    const int propThreads = EE * 32;
    const int propBlocks = (propThreads + 255) / 256;

    // full ChebConv matmul chain: OUT = sum_k T_k(X) @ W[k]
    auto chebMM = [&](const float* X, const __bf16* WpBase, size_t slice, int C, float* OUT,
                      const int* src, const int* dst) {
        // k = 0 : T0 = X
        gemm(X, WpBase, OUT, C, 0);
        // k = 1 : T1 = prop(X)
        fill_kernel<<<gb256, 256, 0, stream>>>(B1, nd, 0.0f);
        prop_kernel<<<propBlocks, 256, 0, stream>>>(src, dst, whatv, X, B1, 1.0f);
        gemm(B1, WpBase + slice, OUT, C, 1);
        // k = 2 : T2 = 2*prop(T1) - T0
        scale_copy_kernel<<<gb256, 256, 0, stream>>>(B2, X, nd, -1.0f);
        prop_kernel<<<propBlocks, 256, 0, stream>>>(src, dst, whatv, B1, B2, 2.0f);
        gemm(B2, WpBase + 2 * slice, OUT, C, 1);
        // k = 3 : T3 = 2*prop(T2) - T1
        scale_copy_kernel<<<gb256, 256, 0, stream>>>(B0, B1, nd, -1.0f);
        prop_kernel<<<propBlocks, 256, 0, stream>>>(src, dst, whatv, B2, B0, 2.0f);
        gemm(B0, WpBase + 3 * slice, OUT, C, 1);
        // k = 4 : T4 = 2*prop(T3) - T2
        scale_copy_kernel<<<gb256, 256, 0, stream>>>(B1, B2, nd, -1.0f);
        prop_kernel<<<propBlocks, 256, 0, stream>>>(src, dst, whatv, B0, B1, 2.0f);
        gemm(B1, WpBase + 4 * slice, OUT, C, 1);
    };

    const int egb = (EE + 255) / 256;
    for (int t = 0; t < T_STEPS; ++t) {
        const int* src = ei + (size_t)t * 2 * EE;
        const int* dst = src + EE;
        const float* ewt = ew + (size_t)t * EE;

        fill_kernel<<<(NN + 255) / 256, 256, 0, stream>>>(deg, NN, 0.0f);
        deg_kernel<<<egb, 256, 0, stream>>>(src, dst, ewt, deg);
        what_kernel<<<egb, 256, 0, stream>>>(src, dst, ewt, deg, whatv);

        for (int l = 0; l < LL; ++l) {
            const float* cur = (l == 0) ? emb : h0;
            float* h = (l == 0) ? h0 : h1;
            const float* bxl = bx + (size_t)l * 384;
            const float* bhl = bh + (size_t)l * 384;

            chebMM(cur, WxP + (size_t)l * KC * sliceX, sliceX, 384, Xacc, src, dst);
            chebMM(h, WhP01 + (size_t)l * KC * slice01, slice01, 256, Hacc, src, dst);
            gate_zr_kernel<<<gb256, 256, 0, stream>>>(Xacc, Hacc, bxl, bhl, h, Zbuf, HRbuf);
            chebMM(HRbuf, WhP2 + (size_t)l * KC * slice2, slice2, 128, Hacc, src, dst);
            gate_h_kernel<<<gb256, 256, 0, stream>>>(Xacc, Hacc, bxl, bhl, Zbuf, h);
        }
    }

    head_kernel<<<(NN + 255) / 256, 256, 0, stream>>>(h1, Wd, bd, out);
    hipMemcpyAsync(out + (size_t)NN * 16, h0, ND * 4, hipMemcpyDeviceToDevice, stream);
    hipMemcpyAsync(out + (size_t)NN * 16 + ND, h1, ND * 4, hipMemcpyDeviceToDevice, stream);
}
